// SelfAttention_25469156065310
// MI455X (gfx1250) — matrix operand order repaired and checked by reading, not compiled, not run on hardware
//
#include <hip/hip_runtime.h>
#include <hip/hip_bf16.h>

// Problem constants (match reference)
#define BATCH  4
#define SEQ    2048
#define DMODEL 1024
#define NHEAD  16
#define DHEAD  64
#define TDIM   (3 * DMODEL)   // 3072
#define MROWS  (BATCH * SEQ)  // 8192

typedef __attribute__((ext_vector_type(16))) __bf16          v16bf;
typedef __attribute__((ext_vector_type(8)))  float           v8f;
typedef __attribute__((ext_vector_type(16))) unsigned short  v16u;
typedef __attribute__((ext_vector_type(4)))  unsigned int    v4u;
typedef __attribute__((ext_vector_type(4)))  int             v4i;
typedef __attribute__((ext_vector_type(4)))  float           v4f;

#define GLOBAL_AS __attribute__((address_space(1)))
#define LDS_AS    __attribute__((address_space(3)))

#if __has_builtin(__builtin_amdgcn_global_load_async_to_lds_b128) && \
    __has_builtin(__builtin_amdgcn_s_wait_asynccnt)
#define USE_ASYNC_LDS 1
#else
#define USE_ASYNC_LDS 0
#endif

__device__ __forceinline__ unsigned short f32_to_bf16(float f) {
    unsigned int u = __float_as_uint(f);
    u += 0x7FFFu + ((u >> 16) & 1u);     // round-to-nearest-even
    return (unsigned short)(u >> 16);
}

// 32-byte bf16 operand: viewable as WMMA vector, ushorts, or two 16B loads
union V16 {
    v16u  u;
    v16bf b;
    v4u   q[2];
};

// ---------------------------------------------------------------------------
// Kernel 0: elementwise f32 -> bf16 (one pass; GEMMs then load pure bf16)
// ---------------------------------------------------------------------------
__global__ __launch_bounds__(256)
void cvt_bf16_kernel(const float* __restrict__ src,
                     unsigned short* __restrict__ dst, int n)
{
    const int i = (blockIdx.x * 256 + threadIdx.x) * 4;
    if (i + 3 < n) {
        v4f f = *(const v4f*)(src + i);
        dst[i + 0] = f32_to_bf16(f.x);
        dst[i + 1] = f32_to_bf16(f.y);
        dst[i + 2] = f32_to_bf16(f.z);
        dst[i + 3] = f32_to_bf16(f.w);
    }
}

// ---------------------------------------------------------------------------
// Kernel 1/3: C[M,N] = A[M,K] @ W[N,K]^T + bias, all operands bf16,
// output either bf16 (qkv) or f32 (final). Per wave: 32(M) x 64(N) tile,
// K-loop in 32-wide chunks -> 8 WMMAs/chunk, B reused 2x, A reused 4x.
// ---------------------------------------------------------------------------
__global__ __launch_bounds__(32)
void gemm_bf16_kernel(const unsigned short* __restrict__ A,   // bf16 [M,K]
                      const unsigned short* __restrict__ W,   // bf16 [N,K]
                      const float* __restrict__ bias,         // [N]
                      unsigned short* __restrict__ OutBf,     // bf16 [M,N] or null
                      float* __restrict__ OutF,               // f32  [M,N] or null
                      const int N, const int K, const int write_bf16)
{
    const int lane = threadIdx.x;
    const int half = lane >> 4;      // 0: lanes 0-15, 1: lanes 16-31
    const int l16  = lane & 15;
    const int m0 = blockIdx.x * 32;
    const int n0 = blockIdx.y * 64;

    v8f acc[2][4] = {};

    const unsigned short* arow0 = A + (size_t)(m0 + l16) * K;
    const unsigned short* arow1 = A + (size_t)(m0 + 16 + l16) * K;

    for (int k0 = 0; k0 < K; k0 += 32) {
        __builtin_prefetch(arow0 + k0 + 256, 0, 0);   // global_prefetch_b8
        __builtin_prefetch(arow1 + k0 + 256, 0, 0);
        // A tiles 16x32 bf16: lane l<16 holds row l, K {0..7,16..23}; l>=16: {8..15,24..31}
        V16 a0, a1;
        a0.q[0] = *(const v4u*)(arow0 + k0 + half * 8);
        a0.q[1] = *(const v4u*)(arow0 + k0 + 16 + half * 8);
        a1.q[0] = *(const v4u*)(arow1 + k0 + half * 8);
        a1.q[1] = *(const v4u*)(arow1 + k0 + 16 + half * 8);
#pragma unroll
        for (int t = 0; t < 4; ++t) {
            // B tile 32x16 bf16: lane = col (W row n0+t*16+l16), K {half*8..+7, 16+half*8..+7}
            const unsigned short* wrow =
                W + (size_t)(n0 + t * 16 + l16) * K + k0 + half * 8;
            V16 b;
            b.q[0] = *(const v4u*)(wrow);
            b.q[1] = *(const v4u*)(wrow + 16);
            acc[0][t] = __builtin_amdgcn_wmma_f32_16x16x32_bf16(
                false, a0.b, false, b.b, (short)0, acc[0][t], false, false);
            acc[1][t] = __builtin_amdgcn_wmma_f32_16x16x32_bf16(
                false, a1.b, false, b.b, (short)0, acc[1][t], false, false);
        }
    }

    // C layout: VGPR r -> row (r + half*8), col l16
#pragma unroll
    for (int mi = 0; mi < 2; ++mi) {
#pragma unroll
        for (int t = 0; t < 4; ++t) {
            const float bval = bias[n0 + t * 16 + l16];
#pragma unroll
            for (int r = 0; r < 8; ++r) {
                const size_t row = (size_t)(m0 + mi * 16 + r + half * 8);
                const size_t idx = row * N + n0 + t * 16 + l16;
                const float v = acc[mi][t][r] + bval;
                if (write_bf16) OutBf[idx] = f32_to_bf16(v);
                else            OutF[idx]  = v;
            }
        }
    }
}

// ---------------------------------------------------------------------------
// Kernel 2: flash attention. One wave per (b, h, 16-query tile).
// qkv bf16 [MROWS, 3D]: Q at col h*64, K at col D + h*64, V at col 2D + h*64.
// ---------------------------------------------------------------------------
__global__ __launch_bounds__(32)
void attention_kernel(const unsigned short* __restrict__ qkv,  // bf16 [MROWS, TDIM]
                      unsigned short* __restrict__ aout,       // bf16 [MROWS, DMODEL]
                      const int* __restrict__ causal_flag)
{
    __shared__ unsigned short Plds[16 * 32];   // P tile, bf16, row-major
    __shared__ unsigned short Vlds[32 * 64];   // V block, bf16, row-major

    const int lane = threadIdx.x;
    const int half = lane >> 4;
    const int l16  = lane & 15;
    const int q0   = blockIdx.x * 16;
    const int bh   = blockIdx.y;
    const int b    = bh / NHEAD;
    const int h    = bh % NHEAD;
    const int causal = causal_flag[0];

    const size_t rowbase = (size_t)b * SEQ;

    // Q as two 16x32 bf16 A-matrices (dh chunks [0,32), [32,64))
    V16 aq[2];
    {
        const unsigned short* qrow =
            qkv + (rowbase + q0 + l16) * TDIM + h * DHEAD;
#pragma unroll
        for (int c = 0; c < 2; ++c) {
            aq[c].q[0] = *(const v4u*)(qrow + c * 32 + half * 8);
            aq[c].q[1] = *(const v4u*)(qrow + c * 32 + 16 + half * 8);
        }
    }

    v8f o[4] = {};
    float mrow[8], lrow[8];
#pragma unroll
    for (int r = 0; r < 8; ++r) { mrow[r] = -1e30f; lrow[r] = 0.0f; }

    const int kv_end = causal ? (q0 + 16) : SEQ;

    for (int k0 = 0; k0 < kv_end; k0 += 32) {
        // ---- kick off V block (32 keys x 64 dh) -> LDS as early as possible
#if USE_ASYNC_LDS
        {
            // builtin signature (per hipcc diagnostic): (int4 AS1*, int4 AS3*, imm, imm)
            GLOBAL_AS v4i* gsrc = (GLOBAL_AS v4i*)(v4i*)(unsigned short*)
                (qkv + (rowbase + k0 + lane) * TDIM + 2 * DMODEL + h * DHEAD);
            LDS_AS v4i* ldst = (LDS_AS v4i*)(v4i*)&Vlds[lane * 64];
#pragma unroll
            for (int j = 0; j < 8; ++j)
                __builtin_amdgcn_global_load_async_to_lds_b128(
                    gsrc + j, ldst + j, 0, 0);
        }
#else
        {
            const v4u* vsrc = (const v4u*)
                (qkv + (rowbase + k0 + lane) * TDIM + 2 * DMODEL + h * DHEAD);
            v4u* vdst = (v4u*)&Vlds[lane * 64];
#pragma unroll
            for (int j = 0; j < 8; ++j) vdst[j] = vsrc[j];
        }
#endif

        // ---- scores: 16 queries x 32 keys = two 16x16 WMMA tiles, K=DH=64
        v8f sc[2];
#pragma unroll
        for (int s = 0; s < 2; ++s) {
            v8f c = {};
#pragma unroll
            for (int ch = 0; ch < 2; ++ch) {
                const unsigned short* krow =
                    qkv + (rowbase + k0 + s * 16 + l16) * TDIM + DMODEL +
                    h * DHEAD + ch * 32 + half * 8;
                V16 bk;
                bk.q[0] = *(const v4u*)(krow);
                bk.q[1] = *(const v4u*)(krow + 16);
                c = __builtin_amdgcn_wmma_f32_16x16x32_bf16(
                    false, aq[ch].b, false, bk.b, (short)0, c, false, false);
            }
            sc[s] = c;
        }

        // scale + causal mask (branchless per-lane select; EXEC stays all-ones)
#pragma unroll
        for (int s = 0; s < 2; ++s) {
#pragma unroll
            for (int r = 0; r < 8; ++r) {
                float v = sc[s][r] * 0.125f;   // 1/sqrt(64)
                const int qg = q0 + r + half * 8;
                const int kg = k0 + s * 16 + l16;
                if (causal && (kg > qg)) v = -1e30f;
                sc[s][r] = v;
            }
        }

        // ---- online softmax (rows live in 16-lane halves of the wave)
#pragma unroll
        for (int r = 0; r < 8; ++r) {
            float vmax = fmaxf(sc[0][r], sc[1][r]);
#pragma unroll
            for (int off = 1; off < 16; off <<= 1)
                vmax = fmaxf(vmax, __shfl_xor(vmax, off, 32));
            const float mnew = fmaxf(mrow[r], vmax);
            const float corr = __expf(mrow[r] - mnew);
            const float p0 = __expf(sc[0][r] - mnew);
            const float p1 = __expf(sc[1][r] - mnew);
            float rs = p0 + p1;
#pragma unroll
            for (int off = 1; off < 16; off <<= 1)
                rs += __shfl_xor(rs, off, 32);
            lrow[r] = lrow[r] * corr + rs;
            mrow[r] = mnew;
#pragma unroll
            for (int t = 0; t < 4; ++t) o[t][r] *= corr;

            // P -> LDS in C-layout position (row m, cols l16 / 16+l16)
            const int m = r + half * 8;
            Plds[m * 32 + l16]      = f32_to_bf16(p0);
            Plds[m * 32 + 16 + l16] = f32_to_bf16(p1);
        }

#if USE_ASYNC_LDS
        __builtin_amdgcn_s_wait_asynccnt(0);
#endif
        __syncthreads();

        // ---- P re-read from LDS in A-matrix layout (LDS trip = transpose)
        V16 ap;
        ap.q[0] = *(const v4u*)&Plds[l16 * 32 + half * 16];
        ap.q[1] = *(const v4u*)&Plds[l16 * 32 + half * 16 + 8];

        // ---- O += P @ V : 4 dh tiles of 16
#pragma unroll
        for (int t = 0; t < 4; ++t) {
            V16 bv;
#pragma unroll
            for (int j = 0; j < 16; ++j)
                bv.u[j] = Vlds[(half * 16 + j) * 64 + t * 16 + l16];
            o[t] = __builtin_amdgcn_wmma_f32_16x16x32_bf16(
                false, ap.b, false, bv.b, (short)0, o[t], false, false);
        }
        __syncthreads();
    }

    // ---- normalize + store bf16 [b, q, h*64+dh]
#pragma unroll
    for (int r = 0; r < 8; ++r) {
        const float inv = 1.0f / lrow[r];
        const int m = r + half * 8;
        unsigned short* orow =
            aout + (rowbase + q0 + m) * DMODEL + h * DHEAD;
#pragma unroll
        for (int t = 0; t < 4; ++t)
            orow[t * 16 + l16] = f32_to_bf16(o[t][r] * inv);
    }
}

// ---------------------------------------------------------------------------
extern "C" void kernel_launch(void* const* d_in, const int* in_sizes, int n_in,
                              void* d_out, int out_size, void* d_ws, size_t ws_size,
                              hipStream_t stream) {
    const float* x      = (const float*)d_in[0];  // [B,S,D]
    const float* w_in   = (const float*)d_in[1];  // [3D,D]
    const float* b_in   = (const float*)d_in[2];  // [3D]
    const float* w_out  = (const float*)d_in[3];  // [D,D]
    const float* b_out  = (const float*)d_in[4];  // [D]
    const int*   causal = (const int*)d_in[5];    // scalar
    float* out = (float*)d_out;

    // Workspace layout (bf16 buffers), 88 MB total:
    unsigned short* qkv_bf  = (unsigned short*)d_ws;                   // 48 MB
    unsigned short* aout_bf = qkv_bf  + (size_t)MROWS * TDIM;          // 16 MB
    unsigned short* x_bf    = aout_bf + (size_t)MROWS * DMODEL;        // 16 MB
    unsigned short* win_bf  = x_bf    + (size_t)MROWS * DMODEL;        //  6 MB
    unsigned short* wout_bf = win_bf  + (size_t)TDIM  * DMODEL;        //  2 MB

    // 0) one-shot conversions to bf16
    const int n_x  = MROWS * DMODEL;        // 8388608
    const int n_wi = TDIM * DMODEL;         // 3145728
    const int n_wo = DMODEL * DMODEL;       // 1048576
    cvt_bf16_kernel<<<n_x  / 1024, 256, 0, stream>>>(x,     x_bf,    n_x);
    cvt_bf16_kernel<<<n_wi / 1024, 256, 0, stream>>>(w_in,  win_bf,  n_wi);
    cvt_bf16_kernel<<<n_wo / 1024, 256, 0, stream>>>(w_out, wout_bf, n_wo);

    // 1) fused QKV projection (bf16 out)
    dim3 g1(MROWS / 32, TDIM / 64);               // 256 x 48
    gemm_bf16_kernel<<<g1, 32, 0, stream>>>(x_bf, win_bf, b_in,
                                            qkv_bf, nullptr, TDIM, DMODEL, 1);

    // 2) flash attention per (q-tile, b*h)
    dim3 g2(SEQ / 16, BATCH * NHEAD);             // 128 x 64
    attention_kernel<<<g2, 32, 0, stream>>>(qkv_bf, aout_bf, causal);

    // 3) output projection (f32 out + bias)
    dim3 g3(MROWS / 32, DMODEL / 64);             // 256 x 16
    gemm_bf16_kernel<<<g3, 32, 0, stream>>>(aout_bf, wout_bf, b_out,
                                            nullptr, out, DMODEL, DMODEL, 0);
}
